// VectorQuantizer_30365418783153
// MI455X (gfx1250) — compile-verified, matrix operand order checked by reading, not run on hardware
//
#include <hip/hip_runtime.h>
#include <hip/hip_bf16.h>

// ---------------------------------------------------------------------------
// VQ-VAE vector quantizer for MI455X (gfx1250, wave32, WMMA).
//   z: [64,32,32,256] fp32  -> 65536 rows of dim 256
//   E: [1024,256] fp32 codebook
// Outputs (concatenated in d_out, fp32):
//   z_q  [65536*256], encodings [65536], loss [1], perplexity [1]
//
// Distance GEMM runs as bf16 WMMA with hi/lo split (z ~ z_hi + z_lo,
// e ~ e_hi + e_lo; dot ~ hi.hi + hi.lo + lo.hi) -> near-fp32 argmin fidelity
// at 3x the (cheap) matrix cost. ||z||^2 is dropped (argmin-invariant).
// Three independent WMMA accumulator chains avoid D->C serial stalls.
// ---------------------------------------------------------------------------

typedef __attribute__((ext_vector_type(16))) __bf16 v16bf;
typedef __attribute__((ext_vector_type(8)))  float  v8f;

#define N_E      1024
#define E_DIM    256
#define N_TILES  (N_E / 16)    // 64 column tiles of 16 embeddings
#define K_STEPS  (E_DIM / 32)  // 8 K-chunks of 32 for 16x16x32 bf16 WMMA
#define NZ       65536         // 64*32*32 z-vectors
#define ZQ_ELEMS (NZ * E_DIM)

union BF16Frag {
    v16bf          v;
    unsigned short u[16];
    uint4          q[2];
};

__device__ __forceinline__ unsigned short f2bf_rne(float f) {
    unsigned int x = __float_as_uint(f);
    unsigned int r = x + 0x7FFFu + ((x >> 16) & 1u);  // round-to-nearest-even
    return (unsigned short)(r >> 16);
}
__device__ __forceinline__ float bf2f(unsigned short h) {
    return __uint_as_float(((unsigned int)h) << 16);
}

// ---------------------------------------------------------------------------
// Pre-pack codebook into bf16 hi/lo B-fragments.
// ISA 16-bit B 32x16 layout: lane l -> column n = l&15; K = kb..kb+7
// (elems 0-7) and kb+16..kb+23 (elems 8-15) with kb = (l>>4)*8.
// Fragment for (tile t, kstep s, lane l) stored contiguously: 16 halves (32 B).
// ---------------------------------------------------------------------------
__global__ void vq_pack_b(const float* __restrict__ emb,
                          unsigned short* __restrict__ bhi,
                          unsigned short* __restrict__ blo) {
    const int t    = blockIdx.x;          // 0..63
    const int s    = threadIdx.x >> 5;    // 0..7 (wave within block)
    const int lane = threadIdx.x & 31;
    const int n    = t * 16 + (lane & 15);
    const int kb   = s * 32 + (lane >> 4) * 8;

    const float4* src = reinterpret_cast<const float4*>(emb + (size_t)n * E_DIM + kb);
    float4 f0 = src[0], f1 = src[1];   // K = kb .. kb+7
    float4 f2 = src[4], f3 = src[5];   // K = kb+16 .. kb+23
    float vals[16] = { f0.x, f0.y, f0.z, f0.w, f1.x, f1.y, f1.z, f1.w,
                       f2.x, f2.y, f2.z, f2.w, f3.x, f3.y, f3.z, f3.w };

    size_t off = ((size_t)(t * K_STEPS + s) * 32 + lane) * 16;
#pragma unroll
    for (int e = 0; e < 16; ++e) {
        unsigned short h = f2bf_rne(vals[e]);
        bhi[off + e] = h;
        blo[off + e] = f2bf_rne(vals[e] - bf2f(h));  // residual term
    }
}

// ---------------------------------------------------------------------------
// ||e_n||^2, zero histogram + loss accumulator (ws is poisoned each run).
// ---------------------------------------------------------------------------
__global__ void vq_prep(const float* __restrict__ emb,
                        float* __restrict__ enorm,
                        unsigned int* __restrict__ counts,
                        float* __restrict__ loss_accum) {
    const int n = blockIdx.x * blockDim.x + threadIdx.x;  // 0..1023
    const float4* row = reinterpret_cast<const float4*>(emb + (size_t)n * E_DIM);
    float s = 0.f;
#pragma unroll 4
    for (int i = 0; i < E_DIM / 4; ++i) {
        float4 v = row[i];
        s += v.x * v.x + v.y * v.y + v.z * v.z + v.w * v.w;
    }
    enorm[n]  = s;
    counts[n] = 0u;
    if (n == 0) *loss_accum = 0.f;
}

// ---------------------------------------------------------------------------
// Main kernel: each wave owns 16 z-rows, sweeps all 1024 embeddings.
// ---------------------------------------------------------------------------
__launch_bounds__(256)
__global__ void vq_main(const float* __restrict__ z,
                        const float* __restrict__ emb,
                        const unsigned short* __restrict__ bhi,
                        const unsigned short* __restrict__ blo,
                        const float* __restrict__ enorm,
                        float* __restrict__ zq,
                        float* __restrict__ enc,
                        unsigned int* __restrict__ counts,
                        float* __restrict__ loss_accum) {
    const int  lane    = threadIdx.x & 31;
    const int  wave    = threadIdx.x >> 5;
    const int  m       = lane & 15;    // A row / B column owned by this lane
    const int  hi      = lane >> 4;    // K-half selector
    const long rowBase = ((long)blockIdx.x * 8 + wave) * 16;

    // ---- Load A (16 rows x 256) once, split into bf16 hi/lo fragments ----
    // 16-bit A 16x32 layout mirrors B: lane l holds row m=l&15,
    // K = kb..kb+7 (elems 0-7) and kb+16..kb+23 (elems 8-15), kb=(l>>4)*8.
    BF16Frag ahi[K_STEPS], alo[K_STEPS];
    const float* zrow = z + (rowBase + m) * E_DIM;
#pragma unroll
    for (int s = 0; s < K_STEPS; ++s) {
        const int kb = s * 32 + hi * 8;
        const float4* src = reinterpret_cast<const float4*>(zrow + kb);
        float4 f0 = src[0], f1 = src[1], f2 = src[4], f3 = src[5];
        float vals[16] = { f0.x, f0.y, f0.z, f0.w, f1.x, f1.y, f1.z, f1.w,
                           f2.x, f2.y, f2.z, f2.w, f3.x, f3.y, f3.z, f3.w };
#pragma unroll
        for (int e = 0; e < 16; ++e) {
            unsigned short h = f2bf_rne(vals[e]);
            ahi[s].u[e] = h;
            alo[s].u[e] = f2bf_rne(vals[e] - bf2f(h));
        }
    }

    float mind[8];
    int   midx[8];
#pragma unroll
    for (int r = 0; r < 8; ++r) { mind[r] = 3.4e38f; midx[r] = 0; }

    const uint4* bhq = reinterpret_cast<const uint4*>(bhi) + lane * 2;
    const uint4* blq = reinterpret_cast<const uint4*>(blo) + lane * 2;

#pragma unroll 1
    for (int t = 0; t < N_TILES; ++t) {
        if (t + 1 < N_TILES) {  // pull next tile's fragments toward the WGP
            __builtin_prefetch(bhq + (size_t)(t + 1) * K_STEPS * 64, 0, 1);
            __builtin_prefetch(blq + (size_t)(t + 1) * K_STEPS * 64, 0, 1);
        }
        // Three independent accumulator chains: no WMMA->WMMA D->C stalls.
        v8f acc0 = { 0.f, 0.f, 0.f, 0.f, 0.f, 0.f, 0.f, 0.f };
        v8f acc1 = acc0;
        v8f acc2 = acc0;

        const size_t tb = (size_t)t * K_STEPS * 64;  // uint4 units
        BF16Frag bh, bl;                             // current-step fragments
        bh.q[0] = bhq[tb];     bh.q[1] = bhq[tb + 1];
        bl.q[0] = blq[tb];     bl.q[1] = blq[tb + 1];

#pragma unroll
        for (int s = 0; s < K_STEPS; ++s) {
            BF16Frag nbh, nbl;                       // next-step double buffer
            if (s + 1 < K_STEPS) {
                const size_t fo = tb + (size_t)(s + 1) * 64;
                nbh.q[0] = bhq[fo]; nbh.q[1] = bhq[fo + 1];
                nbl.q[0] = blq[fo]; nbl.q[1] = blq[fo + 1];
            }
            acc0 = __builtin_amdgcn_wmma_f32_16x16x32_bf16(false, ahi[s].v, false, bh.v,
                                                           (short)0, acc0, false, false);
            acc1 = __builtin_amdgcn_wmma_f32_16x16x32_bf16(false, ahi[s].v, false, bl.v,
                                                           (short)0, acc1, false, false);
            acc2 = __builtin_amdgcn_wmma_f32_16x16x32_bf16(false, alo[s].v, false, bh.v,
                                                           (short)0, acc2, false, false);
            bh = nbh;
            bl = nbl;
        }

        // D layout: lane l holds (row = r + 8*(l>>4), col n = tile*16 + (l&15))
        const float en = enorm[t * 16 + m];
        const int   n  = t * 16 + m;
#pragma unroll
        for (int r = 0; r < 8; ++r) {
            const float dv = en - 2.0f * (acc0[r] + acc1[r] + acc2[r]);
            if (dv < mind[r]) { mind[r] = dv; midx[r] = n; }
        }
    }

    // ---- Min-reduce over the 16 lanes sharing a row (xor bits 0..3) ----
#pragma unroll
    for (int off = 8; off >= 1; off >>= 1) {
#pragma unroll
        for (int r = 0; r < 8; ++r) {
            const float od = __shfl_xor(mind[r], off, 32);
            const int   oi = __shfl_xor(midx[r], off, 32);
            if (od < mind[r] || (od == mind[r] && oi < midx[r])) {
                mind[r] = od; midx[r] = oi;  // first-occurrence tie-break
            }
        }
    }

    // ---- encodings + histogram (lane 0 -> rows 0-7, lane 16 -> rows 8-15) --
    if (m == 0) {
#pragma unroll
        for (int r = 0; r < 8; ++r) {
            const long row = rowBase + hi * 8 + r;
            enc[row] = (float)midx[r];
            atomicAdd(&counts[midx[r]], 1u);
        }
    }

    // ---- z_q = E[idx] (fp32 gather) + loss partial sum --------------------
    float lsum = 0.f;
#pragma unroll 1
    for (int r = 0; r < 8; ++r) {
#pragma unroll
        for (int h = 0; h < 2; ++h) {
            const int  idxn = __shfl(midx[r], h * 16, 32);  // owner half-wave
            const long row  = rowBase + h * 8 + r;
            const float4* es = reinterpret_cast<const float4*>(emb + (size_t)idxn * E_DIM);
            const float4* zs = reinterpret_cast<const float4*>(z + row * E_DIM);
            float4*       dq = reinterpret_cast<float4*>(zq + row * E_DIM);
#pragma unroll
            for (int j = 0; j < 2; ++j) {
                const int c = lane + j * 32;  // 64 float4 per row
                float4 ev = es[c];
                float4 zv = zs[c];
                dq[c] = ev;
                const float dx = ev.x - zv.x, dy = ev.y - zv.y;
                const float dz = ev.z - zv.z, dw = ev.w - zv.w;
                lsum += dx * dx + dy * dy + dz * dz + dw * dw;
            }
        }
    }
#pragma unroll
    for (int off = 16; off >= 1; off >>= 1) lsum += __shfl_xor(lsum, off, 32);
    if (lane == 0) atomicAdd(loss_accum, lsum);
}

// ---------------------------------------------------------------------------
// loss = 1.25 * sum/(NZ*E_DIM);  perplexity = exp(-sum p log(p+1e-10))
// ---------------------------------------------------------------------------
__global__ void vq_finalize(const unsigned int* __restrict__ counts,
                            const float* __restrict__ loss_accum,
                            float* __restrict__ out) {
    __shared__ float sm[1024];
    const int tid = threadIdx.x;
    const float p = (float)counts[tid] * (1.0f / (float)NZ);
    sm[tid] = -p * logf(p + 1e-10f);
    __syncthreads();
    for (int s = 512; s > 0; s >>= 1) {
        if (tid < s) sm[tid] += sm[tid + s];
        __syncthreads();
    }
    if (tid == 0) {
        out[ZQ_ELEMS + NZ]     = 1.25f * (*loss_accum) * (1.0f / (float)ZQ_ELEMS);
        out[ZQ_ELEMS + NZ + 1] = expf(sm[0]);
    }
}

// ---------------------------------------------------------------------------
extern "C" void kernel_launch(void* const* d_in, const int* in_sizes, int n_in,
                              void* d_out, int out_size, void* d_ws, size_t ws_size,
                              hipStream_t stream) {
    const float* z   = (const float*)d_in[0];
    const float* emb = (const float*)d_in[1];

    char* ws = (char*)d_ws;
    unsigned short* bhi        = (unsigned short*)(ws);                     // 512 KB
    unsigned short* blo        = (unsigned short*)(ws + 512 * 1024);        // 512 KB
    float*          enorm      = (float*)(ws + 1024 * 1024);                // 4 KB
    unsigned int*   counts     = (unsigned int*)(ws + 1024 * 1024 + 4096);  // 4 KB
    float*          loss_accum = (float*)(ws + 1024 * 1024 + 8192);         // 4 B

    float* out = (float*)d_out;
    float* zq  = out;              // [NZ*E_DIM]
    float* enc = out + ZQ_ELEMS;   // [NZ]

    vq_pack_b<<<N_TILES, 256, 0, stream>>>(emb, bhi, blo);
    vq_prep<<<4, 256, 0, stream>>>(emb, enorm, counts, loss_accum);
    vq_main<<<NZ / 128, 256, 0, stream>>>(z, emb, bhi, blo, enorm,
                                          zq, enc, counts, loss_accum);
    vq_finalize<<<1, 1024, 0, stream>>>(counts, loss_accum, out);
}
